// lstmCell_19421842112787
// MI455X (gfx1250) — matrix operand order repaired and checked by reading, not compiled, not run on hardware
//
#include <hip/hip_runtime.h>

// ---------------------------------------------------------------------------
// LSTM on MI455X (gfx1250): bf16 WMMA 16x16x32, fp32 accum, fp32 cell state.
// B-tiles staged to LDS via async global->LDS DMA (double-buffered) when the
// toolchain exposes the gfx1250 async builtins; falls back to load+ds_store.
// ---------------------------------------------------------------------------

typedef __attribute__((ext_vector_type(16))) __bf16         v16bf;
typedef __attribute__((ext_vector_type(8)))  float          v8f;
typedef __attribute__((ext_vector_type(8)))  unsigned short v8us;
typedef __attribute__((ext_vector_type(16))) unsigned short v16us;
typedef __attribute__((ext_vector_type(4)))  int            v4i;

typedef __attribute__((address_space(1))) v4i* gv4i_p;  // global int4*
typedef __attribute__((address_space(3))) v4i* lv4i_p;  // LDS int4*

constexpr int B_   = 128;
constexpr int S_   = 512;
constexpr int I_   = 256;
constexpr int H_   = 1024;
constexpr int C_   = 10;
constexpr int KTOT = I_ + H_;   // 1280 (fused [x | h] contraction dim)
constexpr int KIT  = KTOT / 32; // 40 K-iterations

#if __has_builtin(__builtin_amdgcn_global_load_async_to_lds_b128) && \
    __has_builtin(__builtin_amdgcn_s_wait_asynccnt)
#define USE_ASYNC_LDS 1
#else
#define USE_ASYNC_LDS 0
#endif

__device__ __forceinline__ unsigned short f2bf(float f) {
  unsigned u = __float_as_uint(f);
  u += 0x7fffu + ((u >> 16) & 1u);        // round-to-nearest-even
  return (unsigned short)(u >> 16);
}

__device__ __forceinline__ float sigmoidf_(float x) {
  return 1.0f / (1.0f + __expf(-x));
}

// branch-free tanh: 1 - 2/(e^{2x}+1); saturates correctly at +/-inf
__device__ __forceinline__ float tanhf_(float x) {
  return 1.0f - 2.0f / (__expf(2.0f * x) + 1.0f);
}

// ---------------- one-time prep kernels ----------------

__global__ void cvt_x_bf16(const float* __restrict__ x,
                           unsigned short* __restrict__ xbf, int n) {
  int i = blockIdx.x * blockDim.x + threadIdx.x;
  if (i < n) xbf[i] = f2bf(x[i]);
}

// W_cat[g][k] : k<256 -> W_ih[g][k], else W_hh[g][k-256]  (4096 x 1280, bf16)
__global__ void build_wcat(const float* __restrict__ Wih,
                           const float* __restrict__ Whh,
                           unsigned short* __restrict__ Wcat, int n) {
  int i = blockIdx.x * blockDim.x + threadIdx.x;
  if (i >= n) return;
  int g = i / KTOT, k = i - g * KTOT;
  float v = (k < I_) ? Wih[(size_t)g * I_ + k] : Whh[(size_t)g * H_ + (k - I_)];
  Wcat[i] = f2bf(v);
}

__global__ void bias_sum(const float* __restrict__ bih,
                         const float* __restrict__ bhh,
                         float* __restrict__ bsum, int n) {
  int i = blockIdx.x * blockDim.x + threadIdx.x;
  if (i < n) bsum[i] = bih[i] + bhh[i];
}

__global__ void init_hc(const float* __restrict__ h0, const float* __restrict__ c0,
                        unsigned short* __restrict__ hbf,
                        float* __restrict__ hf32, float* __restrict__ cbuf, int n) {
  int i = blockIdx.x * blockDim.x + threadIdx.x;
  if (i >= n) return;
  hbf[i]  = f2bf(h0[i]);
  hf32[i] = h0[i];
  cbuf[i] = c0[i];
}

// ---------------- per-timestep fused kernel ----------------
// grid.x = H/16 (column tile of h), block = 256 (8 waves).
// Wave w computes rows [16w,16w+16) x 16 cols for all four gates,
// then does the elementwise cell update.

__global__ __launch_bounds__(256)
void lstm_step(int s,
               const unsigned short* __restrict__ xbf,   // (B,S,I) bf16
               const unsigned short* __restrict__ Wcat,  // (4H, 1280) bf16
               const float* __restrict__ bsum,           // (4H,)
               const unsigned short* __restrict__ hprev, // (B,H) bf16
               unsigned short* __restrict__ hnext,       // (B,H) bf16
               float* __restrict__ cbuf,                 // (B,H) f32
               float* __restrict__ hf32) {               // (B,H) f32
  // [buf][gate*16+col][k] : 4 gate tiles of 32(K) x 16(N) bf16, double buffered
  __shared__ unsigned short ldsB[2][4 * 16 * 32];

  const int tid   = threadIdx.x;
  const int wave  = tid >> 5;
  const int lane  = tid & 31;
  const int lhalf = lane >> 4;   // 0: lanes 0-15, 1: lanes 16-31
  const int lmod  = lane & 15;
  const int j0    = blockIdx.x * 16;

  v8f acc[4] = {};               // i, f, g, o accumulators (16x16 f32 each)

  const int rowA = wave * 16 + lmod;  // batch row this lane supplies for A
  const unsigned short* xrow = xbf   + (size_t)rowA * (S_ * I_) + (size_t)s * I_;
  const unsigned short* hrow = hprev + (size_t)rowA * H_;

  // staging role of this thread: gate q, column n, 8-half chunk `part`
  const int sq    = tid >> 6;          // 0..3
  const int sn    = (tid >> 2) & 15;   // 0..15
  const int spart = tid & 3;           // 0..3
  const unsigned short* wsrc = Wcat + (size_t)(sq * H_ + j0 + sn) * KTOT + spart * 8;
  const int ldoff = (sq * 16 + sn) * 32 + spart * 8;

#if USE_ASYNC_LDS
  // async DMA one b128 chunk of the W tile for K-iter `it` into buffer `buf`
  auto stage = [&](int it, int buf) {
    __builtin_amdgcn_global_load_async_to_lds_b128(
        (gv4i_p)(wsrc + it * 32),
        (lv4i_p)&ldsB[buf][ldoff],
        /*offset=*/0, /*cpol=*/0);
  };
  stage(0, 0);
  __builtin_amdgcn_s_wait_asynccnt(0);
  __syncthreads();
#endif

  for (int it = 0; it < KIT; ++it) {
    const int K0  = it * 32;
    const int cur = it & 1;

#if USE_ASYNC_LDS
    if (it + 1 < KIT) stage(it + 1, (it + 1) & 1);
#else
    {
      const v8us w = *(const v8us*)(wsrc + K0);
      *(v8us*)(&ldsB[0][ldoff]) = w;
    }
    __syncthreads();
#endif

    // ---- A fragment: 16-bit A 16x32 layout (lane L<16: K 0-7 & 16-23) ----
    const int k0 = lhalf * 8;
    v8us alo, ahi;
    if (K0 < I_) {
      alo = *(const v8us*)(xrow + K0 + k0);
      ahi = *(const v8us*)(xrow + K0 + k0 + 16);
    } else {
      alo = *(const v8us*)(hrow + (K0 - I_) + k0);
      ahi = *(const v8us*)(hrow + (K0 - I_) + k0 + 16);
    }
    v16us afull = __builtin_shufflevector(alo, ahi,
        0, 1, 2, 3, 4, 5, 6, 7, 8, 9, 10, 11, 12, 13, 14, 15);
    v16bf a = __builtin_bit_cast(v16bf, afull);

    // ---- 4 gate WMMAs, B fragments read from LDS ----
#pragma unroll
    for (int q = 0; q < 4; ++q) {
      const unsigned short* bp =
          &ldsB[USE_ASYNC_LDS ? cur : 0][(q * 16 + lmod) * 32 + lhalf * 8];
      v8us blo = *(const v8us*)(bp);
      v8us bhi = *(const v8us*)(bp + 16);
      v16us bfull = __builtin_shufflevector(blo, bhi,
          0, 1, 2, 3, 4, 5, 6, 7, 8, 9, 10, 11, 12, 13, 14, 15);
      v16bf b = __builtin_bit_cast(v16bf, bfull);
      acc[q] = __builtin_amdgcn_wmma_f32_16x16x32_bf16(
          /*neg_a=*/false, a, /*neg_b=*/false, b,
          /*c_mod=*/(short)0, acc[q], /*reuse_a=*/false, /*reuse_b=*/false);
    }

#if USE_ASYNC_LDS
    __builtin_amdgcn_s_wait_asynccnt(0);
#endif
    __syncthreads();
  }

  // ---- bias + nonlinearities + cell update (C/D layout: col = lane&15,
  //      row = vgpr + 8*(lane>>4)) ----
  const int col = j0 + lmod;
  const float bi = bsum[0 * H_ + col];
  const float bf = bsum[1 * H_ + col];
  const float bg = bsum[2 * H_ + col];
  const float bo = bsum[3 * H_ + col];

#pragma unroll
  for (int v = 0; v < 8; ++v) {
    const int row    = wave * 16 + v + 8 * lhalf;
    const size_t idx = (size_t)row * H_ + col;
    float iv = sigmoidf_(acc[0][v] + bi);
    float fv = sigmoidf_(acc[1][v] + bf);
    float gv = tanhf_(acc[2][v] + bg);
    float ov = sigmoidf_(acc[3][v] + bo);
    float cn = fv * cbuf[idx] + iv * gv;
    float hn = ov * tanhf_(cn);
    cbuf[idx]  = cn;
    hf32[idx]  = hn;
    hnext[idx] = f2bf(hn);
  }
}

// ---------------- output projection (tiny: 128x10x1024) ----------------

__global__ void out_proj(const float* __restrict__ hf32,
                         const float* __restrict__ Wout,
                         const float* __restrict__ bout,
                         float* __restrict__ out) {
  int t = blockIdx.x * blockDim.x + threadIdx.x;
  if (t >= B_ * C_) return;
  int b = t / C_, cc = t - b * C_;
  const float* hr = hf32 + (size_t)b * H_;
  const float* wr = Wout + (size_t)cc * H_;
  float acc = 0.0f;
#pragma unroll 4
  for (int k = 0; k < H_; ++k) acc = fmaf(hr[k], wr[k], acc);
  out[t] = acc + bout[cc];
}

// ---------------- launcher ----------------

extern "C" void kernel_launch(void* const* d_in, const int* in_sizes, int n_in,
                              void* d_out, int out_size, void* d_ws, size_t ws_size,
                              hipStream_t stream) {
  const float* x    = (const float*)d_in[0];
  const float* h0   = (const float*)d_in[1];
  const float* c0   = (const float*)d_in[2];
  const float* Wih  = (const float*)d_in[3];
  const float* Whh  = (const float*)d_in[4];
  const float* bih  = (const float*)d_in[5];
  const float* bhh  = (const float*)d_in[6];
  const float* Wout = (const float*)d_in[7];
  const float* bout = (const float*)d_in[8];
  float* out = (float*)d_out;

  // workspace carve-up (all 256B aligned)
  char* ws = (char*)d_ws;
  size_t off = 0;
  auto alloc = [&](size_t bytes) {
    char* p = ws + off;
    off = (off + bytes + 255) & ~(size_t)255;
    return p;
  };
  unsigned short* xbf  = (unsigned short*)alloc((size_t)B_ * S_ * I_ * 2);  // 32 MB
  unsigned short* Wcat = (unsigned short*)alloc((size_t)4 * H_ * KTOT * 2); // 10 MB
  float*          bsm  = (float*)alloc((size_t)4 * H_ * 4);
  unsigned short* hbf0 = (unsigned short*)alloc((size_t)B_ * H_ * 2);
  unsigned short* hbf1 = (unsigned short*)alloc((size_t)B_ * H_ * 2);
  float*          cbuf = (float*)alloc((size_t)B_ * H_ * 4);
  float*          hfin = (float*)alloc((size_t)B_ * H_ * 4);

  // prep
  {
    int n = B_ * S_ * I_;
    cvt_x_bf16<<<(n + 255) / 256, 256, 0, stream>>>(x, xbf, n);
  }
  {
    int n = 4 * H_ * KTOT;
    build_wcat<<<(n + 255) / 256, 256, 0, stream>>>(Wih, Whh, Wcat, n);
  }
  {
    int n = 4 * H_;
    bias_sum<<<(n + 255) / 256, 256, 0, stream>>>(bih, bhh, bsm, n);
  }
  {
    int n = B_ * H_;
    init_hc<<<(n + 255) / 256, 256, 0, stream>>>(h0, c0, hbf0, hfin, cbuf, n);
  }

  // recurrence: stream ordering = inter-step barrier; ping-pong h (bf16)
  for (int s = 0; s < S_; ++s) {
    const unsigned short* hp = (s & 1) ? hbf1 : hbf0;
    unsigned short*       hn = (s & 1) ? hbf0 : hbf1;
    lstm_step<<<H_ / 16, 256, 0, stream>>>(s, xbf, Wcat, bsm, hp, hn, cbuf, hfin);
  }

  // classifier head
  {
    int n = B_ * C_;
    out_proj<<<(n + 63) / 64, 64, 0, stream>>>(hfin, Wout, bout, out);
  }
}
